// SchNet_90546500534291
// MI455X (gfx1250) — matrix operand order repaired and checked by reading, not compiled, not run on hardware
//
#include <hip/hip_runtime.h>
#include <math.h>

#define NATOMS 50000
#define NEDGES 1600000
#define NRBF   50
#define NB     64
#define NLAYERS 3
#define CUT    5.0f
#define LOG2F_ 0.69314718056f
#define PI_F   3.14159265358979f

typedef __attribute__((ext_vector_type(16))) _Float16 v16h;
typedef __attribute__((ext_vector_type(8)))  _Float16 v8h;
typedef __attribute__((ext_vector_type(8)))  float    v8f;

union V16H { v16h v; v8h h[2]; _Float16 e[16]; };
union V8F  { v8f v; float e[8]; };

__device__ __forceinline__ float sspf(float x) {
    // numerically stable softplus(x) - log(2)
    return fmaxf(x, 0.f) + __logf(1.f + __expf(-fabsf(x))) - LOG2F_;
}

__device__ __forceinline__ v8f wmma16(v16h a, v16h b, v8f c) {
    return __builtin_amdgcn_wmma_f32_16x16x32_f16(false, a, false, b, (short)0, c,
                                                  false, false);
}

// ---------------------------------------------------------------------------
// x = emb[type] + te @ time_W^T + time_b      (one block of 64 threads / atom)
// ---------------------------------------------------------------------------
__global__ void k_init(const int* __restrict__ types, const float* __restrict__ t,
                       const float* __restrict__ emb, const float* __restrict__ gfp,
                       const float* __restrict__ Wt,  const float* __restrict__ bt,
                       float* __restrict__ x) {
    __shared__ float te[NB];
    int a = blockIdx.x;
    int n = threadIdx.x;
    float tv = t[a];
    float w = (n < 32) ? __sinf(tv * gfp[n] * (2.0f * PI_F))
                       : __cosf(tv * gfp[n - 32] * (2.0f * PI_F));
    te[n] = w;
    __syncthreads();
    float acc = bt[n] + emb[types[a] * NB + n];
    #pragma unroll 8
    for (int f = 0; f < NB; ++f) acc += te[f] * Wt[n * NB + f];
    x[a * NB + n] = acc;
}

// ---------------------------------------------------------------------------
// per-edge distance + cosine cutoff (once, reused by all layers)
// ---------------------------------------------------------------------------
__global__ void k_prep(const float* __restrict__ r, float* __restrict__ de,
                       float* __restrict__ rcut) {
    int e = blockIdx.x * blockDim.x + threadIdx.x;
    if (e >= NEDGES) return;
    float a = r[e * 3 + 0], b = r[e * 3 + 1], c = r[e * 3 + 2];
    float d = sqrtf(a * a + b * b + c * c);
    de[e] = d;
    rcut[e] = (d < CUT) ? 0.5f * (__cosf(d * (PI_F / CUT)) + 1.f) : 0.f;
}

__global__ void k_zero(float* __restrict__ p, int n) {
    int i = blockIdx.x * blockDim.x + threadIdx.x;
    if (i < n) p[i] = 0.f;
}

// ---------------------------------------------------------------------------
// h = x @ W^T   (N x 64) * (64 x 64), WMMA, 16 atoms per wave
// ---------------------------------------------------------------------------
__global__ void k_in2f(const float* __restrict__ x, const float* __restrict__ W,
                       float* __restrict__ h) {
    __shared__ __align__(32) _Float16 Wl[NB * NB];   // B layout: [n][k]
    for (int i = threadIdx.x; i < NB * NB; i += blockDim.x) {
        int n = i >> 6, k = i & 63;
        Wl[i] = (_Float16)W[n * NB + k];
    }
    __syncthreads();
    int lane = threadIdx.x & 31;
    int hl = lane >> 4, m = lane & 15;
    int wid = (blockIdx.x * blockDim.x + threadIdx.x) >> 5;
    int nw  = (gridDim.x * blockDim.x) >> 5;
    for (int tile = wid; tile < NATOMS / 16; tile += nw) {
        int a0 = tile * 16;
        V16H a[2];
        #pragma unroll
        for (int s = 0; s < 2; ++s) {
            const float* p = x + (a0 + m) * NB + s * 32 + (hl << 3);
            #pragma unroll
            for (int q = 0; q < 8; ++q) {
                a[s].e[q]     = (_Float16)p[q];
                a[s].e[q + 8] = (_Float16)p[q + 16];
            }
        }
        #pragma unroll
        for (int t = 0; t < 4; ++t) {
            v8f c = {};
            #pragma unroll
            for (int s = 0; s < 2; ++s) {
                V16H b; const _Float16* pb = Wl + (t * 16 + m) * NB + s * 32 + (hl << 3);
                b.h[0] = *(const v8h*)(pb); b.h[1] = *(const v8h*)(pb + 16);
                c = wmma16(a[s].v, b.v, c);
            }
            V8F cc; cc.v = c;
            int n = t * 16 + m;
            #pragma unroll
            for (int r = 0; r < 8; ++r)
                h[(a0 + (hl << 3) + r) * NB + n] = cc.e[r];
        }
    }
}

// ---------------------------------------------------------------------------
// edge filter MLP + cutoff + gather h[idx_j] + atomic scatter-add to agg
// 16 edges per wave; f_ij synthesized directly into A fragments
// ---------------------------------------------------------------------------
__global__ void k_edge(const float* __restrict__ de, const float* __restrict__ rcut_g,
                       const int* __restrict__ idx_i, const int* __restrict__ idx_j,
                       const float* __restrict__ h,
                       const float* __restrict__ W1, const float* __restrict__ b1,
                       const float* __restrict__ W2, const float* __restrict__ b2,
                       float* __restrict__ agg) {
    __shared__ __align__(32) _Float16 W1l[NB * NB];      // [n][k(rbf, padded)]
    __shared__ __align__(32) _Float16 W2l[NB * NB];      // [n][k]
    __shared__ __align__(32) _Float16 act[8][16 * NB];   // per-wave 16x64 tile
    __shared__ float b1l[NB], b2l[NB];
    for (int i = threadIdx.x; i < NB * NB; i += blockDim.x) {
        int n = i >> 6, k = i & 63;
        W1l[i] = (k < NRBF) ? (_Float16)W1[n * NRBF + k] : (_Float16)0.f;
        W2l[i] = (_Float16)W2[n * NB + k];
    }
    if (threadIdx.x < NB) { b1l[threadIdx.x] = b1[threadIdx.x]; b2l[threadIdx.x] = b2[threadIdx.x]; }
    __syncthreads();

    int lane = threadIdx.x & 31;
    int hl = lane >> 4, m = lane & 15;
    _Float16* T = &act[threadIdx.x >> 5][0];
    int wid = (blockIdx.x * blockDim.x + threadIdx.x) >> 5;
    int nw  = (gridDim.x * blockDim.x) >> 5;
    const float spacing = CUT / (NRBF - 1);
    const float coeff = -0.5f / (spacing * spacing);

    for (int tile = wid; tile < NEDGES / 16; tile += nw) {
        int e0 = tile * 16;
        float d = de[e0 + m];
        // A = f_ij tile (K = rbf index, zero-padded to 64) built in registers
        V16H a[2];
        #pragma unroll
        for (int s = 0; s < 2; ++s) {
            #pragma unroll
            for (int q = 0; q < 16; ++q) {
                int k = s * 32 + (hl << 3) + ((q & 8) << 1) + (q & 7);
                float f = 0.f;
                if (k < NRBF) { float t0 = d - k * spacing; f = __expf(coeff * t0 * t0); }
                a[s].e[q] = (_Float16)f;
            }
        }
        // GEMM1 + bias + ssp -> LDS act tile (re-layout C->A through LDS)
        #pragma unroll
        for (int t = 0; t < 4; ++t) {
            v8f c = {};
            #pragma unroll
            for (int s = 0; s < 2; ++s) {
                V16H b; const _Float16* pb = W1l + (t * 16 + m) * NB + s * 32 + (hl << 3);
                b.h[0] = *(const v8h*)(pb); b.h[1] = *(const v8h*)(pb + 16);
                c = wmma16(a[s].v, b.v, c);
            }
            V8F cc; cc.v = c;
            int n = t * 16 + m;
            float bias = b1l[n];
            #pragma unroll
            for (int r = 0; r < 8; ++r)
                T[((hl << 3) + r) * NB + n] = (_Float16)sspf(cc.e[r] + bias);
        }
        __builtin_amdgcn_wave_barrier();   // keep LDS write->read ordered
        // reload act in A-fragment layout
        V16H a2[2];
        #pragma unroll
        for (int s = 0; s < 2; ++s) {
            const _Float16* p = T + m * NB + s * 32 + (hl << 3);
            a2[s].h[0] = *(const v8h*)(p);
            a2[s].h[1] = *(const v8h*)(p + 16);
        }
        // per-row edge metadata
        int ii[8], jj[8]; float rcl[8];
        #pragma unroll
        for (int r = 0; r < 8; ++r) {
            int e = e0 + (hl << 3) + r;
            ii[r] = idx_i[e]; jj[r] = idx_j[e]; rcl[r] = rcut_g[e];
        }
        // GEMM2 + bias, * rcut, * h[idx_j], atomic add into agg[idx_i]
        #pragma unroll
        for (int t = 0; t < 4; ++t) {
            v8f c = {};
            #pragma unroll
            for (int s = 0; s < 2; ++s) {
                V16H b; const _Float16* pb = W2l + (t * 16 + m) * NB + s * 32 + (hl << 3);
                b.h[0] = *(const v8h*)(pb); b.h[1] = *(const v8h*)(pb + 16);
                c = wmma16(a2[s].v, b.v, c);
            }
            V8F cc; cc.v = c;
            int n = t * 16 + m;
            float bias = b2l[n];
            #pragma unroll
            for (int r = 0; r < 8; ++r) {
                float val = (cc.e[r] + bias) * rcl[r] * h[jj[r] * NB + n];
                atomicAdd(&agg[ii[r] * NB + n], val);
            }
        }
    }
}

// ---------------------------------------------------------------------------
// x += ssp(agg @ W1^T + b1) @ W2^T + b2      (atom-side output MLP + residual)
// ---------------------------------------------------------------------------
__global__ void k_out(const float* __restrict__ agg,
                      const float* __restrict__ W1, const float* __restrict__ b1,
                      const float* __restrict__ W2, const float* __restrict__ b2,
                      float* __restrict__ x) {
    __shared__ __align__(32) _Float16 W1l[NB * NB];
    __shared__ __align__(32) _Float16 W2l[NB * NB];
    __shared__ __align__(32) _Float16 act[8][16 * NB];
    __shared__ float b1l[NB], b2l[NB];
    for (int i = threadIdx.x; i < NB * NB; i += blockDim.x) {
        int n = i >> 6, k = i & 63;
        W1l[i] = (_Float16)W1[n * NB + k];
        W2l[i] = (_Float16)W2[n * NB + k];
    }
    if (threadIdx.x < NB) { b1l[threadIdx.x] = b1[threadIdx.x]; b2l[threadIdx.x] = b2[threadIdx.x]; }
    __syncthreads();

    int lane = threadIdx.x & 31;
    int hl = lane >> 4, m = lane & 15;
    _Float16* T = &act[threadIdx.x >> 5][0];
    int wid = (blockIdx.x * blockDim.x + threadIdx.x) >> 5;
    int nw  = (gridDim.x * blockDim.x) >> 5;

    for (int tile = wid; tile < NATOMS / 16; tile += nw) {
        int a0 = tile * 16;
        V16H a[2];
        #pragma unroll
        for (int s = 0; s < 2; ++s) {
            const float* p = agg + (a0 + m) * NB + s * 32 + (hl << 3);
            #pragma unroll
            for (int q = 0; q < 8; ++q) {
                a[s].e[q]     = (_Float16)p[q];
                a[s].e[q + 8] = (_Float16)p[q + 16];
            }
        }
        #pragma unroll
        for (int t = 0; t < 4; ++t) {
            v8f c = {};
            #pragma unroll
            for (int s = 0; s < 2; ++s) {
                V16H b; const _Float16* pb = W1l + (t * 16 + m) * NB + s * 32 + (hl << 3);
                b.h[0] = *(const v8h*)(pb); b.h[1] = *(const v8h*)(pb + 16);
                c = wmma16(a[s].v, b.v, c);
            }
            V8F cc; cc.v = c;
            int n = t * 16 + m;
            float bias = b1l[n];
            #pragma unroll
            for (int r = 0; r < 8; ++r)
                T[((hl << 3) + r) * NB + n] = (_Float16)sspf(cc.e[r] + bias);
        }
        __builtin_amdgcn_wave_barrier();
        V16H a2[2];
        #pragma unroll
        for (int s = 0; s < 2; ++s) {
            const _Float16* p = T + m * NB + s * 32 + (hl << 3);
            a2[s].h[0] = *(const v8h*)(p);
            a2[s].h[1] = *(const v8h*)(p + 16);
        }
        #pragma unroll
        for (int t = 0; t < 4; ++t) {
            v8f c = {};
            #pragma unroll
            for (int s = 0; s < 2; ++s) {
                V16H b; const _Float16* pb = W2l + (t * 16 + m) * NB + s * 32 + (hl << 3);
                b.h[0] = *(const v8h*)(pb); b.h[1] = *(const v8h*)(pb + 16);
                c = wmma16(a2[s].v, b.v, c);
            }
            V8F cc; cc.v = c;
            int n = t * 16 + m;
            float bias = b2l[n];
            #pragma unroll
            for (int r = 0; r < 8; ++r) {
                int row = a0 + (hl << 3) + r;
                x[row * NB + n] += cc.e[r] + bias;   // residual
            }
        }
    }
}

// ---------------------------------------------------------------------------
extern "C" void kernel_launch(void* const* d_in, const int* in_sizes, int n_in,
                              void* d_out, int out_size, void* d_ws, size_t ws_size,
                              hipStream_t stream) {
    const int*   types  = (const int*)  d_in[0];
    const float* r_ij   = (const float*)d_in[1];
    const int*   idx_i  = (const int*)  d_in[2];
    const int*   idx_j  = (const int*)  d_in[3];
    const float* t      = (const float*)d_in[4];
    const float* emb    = (const float*)d_in[5];
    const float* in2f_W = (const float*)d_in[6];
    const float* fn_W1  = (const float*)d_in[7];
    const float* fn_b1  = (const float*)d_in[8];
    const float* fn_W2  = (const float*)d_in[9];
    const float* fn_b2  = (const float*)d_in[10];
    const float* out_W1 = (const float*)d_in[11];
    const float* out_b1 = (const float*)d_in[12];
    const float* out_W2 = (const float*)d_in[13];
    const float* out_b2 = (const float*)d_in[14];
    const float* gfp_W  = (const float*)d_in[15];
    const float* time_W = (const float*)d_in[16];
    const float* time_b = (const float*)d_in[17];

    float* x   = (float*)d_out;                  // [N,64] state (output)
    float* h   = (float*)d_ws;                   // [N,64]
    float* agg = h + (size_t)NATOMS * NB;        // [N,64]
    float* de  = agg + (size_t)NATOMS * NB;      // [E]
    float* rc  = de + (size_t)NEDGES;            // [E]

    k_init<<<NATOMS, 64, 0, stream>>>(types, t, emb, gfp_W, time_W, time_b, x);
    k_prep<<<(NEDGES + 255) / 256, 256, 0, stream>>>(r_ij, de, rc);

    for (int l = 0; l < NLAYERS; ++l) {
        k_zero<<<(NATOMS * NB + 255) / 256, 256, 0, stream>>>(agg, NATOMS * NB);
        k_in2f<<<256, 256, 0, stream>>>(x, in2f_W + (size_t)l * NB * NB, h);
        k_edge<<<1024, 256, 0, stream>>>(de, rc, idx_i, idx_j, h,
                                         fn_W1 + (size_t)l * NB * NRBF,
                                         fn_b1 + (size_t)l * NB,
                                         fn_W2 + (size_t)l * NB * NB,
                                         fn_b2 + (size_t)l * NB, agg);
        k_out<<<256, 256, 0, stream>>>(agg,
                                       out_W1 + (size_t)l * NB * NB,
                                       out_b1 + (size_t)l * NB,
                                       out_W2 + (size_t)l * NB * NB,
                                       out_b2 + (size_t)l * NB, x);
    }
}
